// MinGRULayer_80530636800507
// MI455X (gfx1250) — matrix operand order repaired and checked by reading, not compiled, not run on hardware
//
#include <hip/hip_runtime.h>
#include <hip/hip_bf16.h>
#include <math.h>

typedef __attribute__((ext_vector_type(16))) __bf16 v16bf;
typedef __attribute__((ext_vector_type(8)))  float  v8f;
typedef __attribute__((ext_vector_type(4)))  int    v4i;
typedef __attribute__((address_space(1)))    v4i    av4i;   // global
typedef __attribute__((address_space(3)))    v4i    sv4i;   // LDS

#define D_DIM 1024
#define S_LEN 4096
#define BATCH 4
#define M_DIM (BATCH * S_LEN)   // 16384 tokens
#define BM 64
#define BN 64
#define KC 32
#define NKSTEP (D_DIM / KC)     // 32 K-iterations
#define LDS_STRIDE 40           // 32 bf16 + 8 pad = 80 bytes/row (16B multiple)

#define CHUNKS 64
#define CLEN   (S_LEN / CHUNKS) // 64
#define NCH    (BATCH * D_DIM)  // 4096 independent scan channels

#if defined(__AMDGCN__) && __has_builtin(__builtin_amdgcn_global_load_async_to_lds_b128)
#define HAVE_ASYNC_LDS 1
#else
#define HAVE_ASYNC_LDS 0
#endif

union Frag {
  v16bf v;
  uint4 q[2];
};

__device__ __forceinline__ unsigned pack_bf16(float lo, float hi) {
  unsigned a = __float_as_uint(lo);
  unsigned b = __float_as_uint(hi);
  a = (a + 0x7FFFu + ((a >> 16) & 1u)) >> 16;   // round-to-nearest-even
  b = (b + 0x7FFFu + ((b >> 16) & 1u)) >> 16;
  return (a & 0xFFFFu) | (b << 16);
}

__device__ __forceinline__ float sigmoid_f(float x) {
  return 1.0f / (1.0f + __expf(-x));
}
__device__ __forceinline__ float tanh_f(float x) {
  float ax = fabsf(x);
  float e  = __expf(-2.0f * ax);
  float r  = (1.0f - e) / (1.0f + e);
  return copysignf(r, x);
}

// 16-byte tile copy: async global->LDS (ASYNCcnt) if available, else via VGPRs.
__device__ __forceinline__ void copy16(const unsigned short* g, unsigned short* l) {
#if HAVE_ASYNC_LDS
  // AS1 pointer == generic global address; AS3 offset == low 32 bits of
  // the generic LDS address (aperture in the high half). Build the
  // address-space-qualified pointers via integer reinterpret casts.
  av4i* gp = (av4i*)(size_t)g;
  sv4i* lp = (sv4i*)(unsigned)(size_t)l;
  __builtin_amdgcn_global_load_async_to_lds_b128(gp, lp, 0, 0);
#else
  *(uint4*)l = *(const uint4*)g;
#endif
}

__device__ __forceinline__ void wait_async_all() {
#if HAVE_ASYNC_LDS
#if __has_builtin(__builtin_amdgcn_s_wait_asynccnt)
  __builtin_amdgcn_s_wait_asynccnt(0);
#else
  asm volatile("s_wait_asynccnt 0x0" ::: "memory");
#endif
#endif
}

// One-time fp32 -> bf16 conversion (8 elements/thread, rne).
__global__ __launch_bounds__(256) void to_bf16(const float* __restrict__ src,
                                               unsigned short* __restrict__ dst,
                                               int n8) {
  int i = blockIdx.x * 256 + threadIdx.x;
  if (i >= n8) return;
  const float4* p = (const float4*)src + (size_t)i * 2;
  float4 v0 = p[0], v1 = p[1];
  uint4 pk = make_uint4(pack_bf16(v0.x, v0.y), pack_bf16(v0.z, v0.w),
                        pack_bf16(v1.x, v1.y), pack_bf16(v1.z, v1.w));
  ((uint4*)dst)[i] = pk;
}

// Fused g/v/d projection GEMM (bf16 WMMA, f32 accum) + gating epilogue.
// 256-thread block -> 64x64 output tile; 8 waves, each a 16x32 strip for all
// 3 weights (6 accumulators). LDS double-buffered, tiles staged with async
// global->LDS copies; one barrier per K-step.
__global__ __launch_bounds__(256) void fused_gvd_gemm(
    const unsigned short* __restrict__ xbf,   // [M,1024] bf16
    const unsigned short* __restrict__ wbf,   // [3][1024][1024] bf16
    const float* __restrict__ bg, const float* __restrict__ bv,
    const float* __restrict__ bd,
    float* __restrict__ xs_out, float* __restrict__ a_out) {
  // [buffer][slot: 0=x, 1..3=W][row*LDS_STRIDE + k]
  __shared__ __align__(16) unsigned short sm[2][4][BM * LDS_STRIDE];

  const int tid  = threadIdx.x;
  const int lane = tid & 31;
  const int wave = tid >> 5;
  const int wm   = wave & 3;   // wave row (4 x 16 = BM)
  const int wn   = wave >> 2;  // wave col (2 x 32 = BN)
  const int bm0  = blockIdx.x * BM;
  const int bn0  = blockIdx.y * BN;

  // staging map: 64 rows x 4 segs of 16B per slot; each thread = 1 chunk/slot
  const int crow = tid >> 2;
  const int cseg = (tid & 3) * 8;          // bf16 element offset

  const size_t xrow = (size_t)(bm0 + crow) * D_DIM + cseg;
  const size_t wrow = (size_t)(bn0 + crow) * D_DIM + cseg;
  const int    loff = crow * LDS_STRIDE + cseg;

  v8f zero = {};
  v8f acc[3][2];
#pragma unroll
  for (int w = 0; w < 3; ++w)
#pragma unroll
    for (int n = 0; n < 2; ++n) acc[w][n] = zero;

  // fragment addresses (constant across K-steps)
  const int ar  = wm * 16 + (lane & 15);
  const int akb = (lane >> 4) * 8;
  const int aoff = ar * LDS_STRIDE + akb;
  const int bc  = wn * 32 + (lane & 15);
  const int bkb = (lane >> 4) * 8;
  const int boff = bc * LDS_STRIDE + bkb;

  // prologue: stage K-step 0 into buffer 0
  copy16(xbf + xrow, &sm[0][0][loff]);
#pragma unroll
  for (int w = 0; w < 3; ++w)
    copy16(wbf + ((size_t)w << 20) + wrow, &sm[0][1 + w][loff]);
  wait_async_all();
  __syncthreads();

  for (int kci = 0; kci < NKSTEP; ++kci) {
    const int cur = kci & 1;
    // prefetch next K-step into the other buffer (safe: barrier of the
    // previous iteration guarantees everyone is done reading it)
    if (kci + 1 < NKSTEP) {
      const int nxt = cur ^ 1;
      const int kc  = (kci + 1) * KC;
      copy16(xbf + xrow + kc, &sm[nxt][0][loff]);
#pragma unroll
      for (int w = 0; w < 3; ++w)
        copy16(wbf + ((size_t)w << 20) + wrow + kc, &sm[nxt][1 + w][loff]);
    }

    // A fragment: 16x32 bf16 (lane l: row l%16, K-half by l/16)
    Frag af;
    af.q[0] = *(const uint4*)&sm[cur][0][aoff];
    af.q[1] = *(const uint4*)&sm[cur][0][aoff + 16];
#pragma unroll
    for (int w = 0; w < 3; ++w) {
#pragma unroll
      for (int n = 0; n < 2; ++n) {
        // B fragment: 32x16 bf16 (lane l: col l%16, K 8h..8h+7 and 16+8h..16+8h+7, h=l/16)
        Frag bfr;
        bfr.q[0] = *(const uint4*)&sm[cur][1 + w][boff + n * 16 * LDS_STRIDE];
        bfr.q[1] = *(const uint4*)&sm[cur][1 + w][boff + n * 16 * LDS_STRIDE + 16];
        acc[w][n] = __builtin_amdgcn_wmma_f32_16x16x32_bf16(
            false, af.v, false, bfr.v, (short)0, acc[w][n], false, false);
      }
    }

    wait_async_all();      // next tile landed in LDS (this wave's copies)
    __syncthreads();       // ...and everyone else's; current tile free
  }

  // epilogue: C layout — VGPR i, lane l: M = i + 8*(l/16), N = l%16
  const int mrow = bm0 + wm * 16 + ((lane >> 4) << 3);
#pragma unroll
  for (int n = 0; n < 2; ++n) {
    const int col = bn0 + wn * 32 + n * 16 + (lane & 15);
    const float bgv = bg[col], bvv = bv[col], bdv = bd[col];
#pragma unroll
    for (int i = 0; i < 8; ++i) {
      const size_t idx = (size_t)(mrow + i) * D_DIM + col;
      float g = acc[0][n][i] + bgv;
      float v = acc[1][n][i] + bvv;
      float d = acc[2][n][i] + bdv;
      xs_out[idx] = sigmoid_f(g) * tanh_f(v);
      a_out[idx]  = 0.001f + 0.998f * sigmoid_f(d);
    }
  }
}

// ---- chunked parallel scan: h_t = a_t*h_{t-1} + x_t --------------------

__global__ __launch_bounds__(256) void scan_pass1(
    const float* __restrict__ xs, const float* __restrict__ a,
    float* __restrict__ Aprod, float* __restrict__ Hend) {
  const int c = blockIdx.x * 256 + threadIdx.x;  // channel 0..4095
  const int k = blockIdx.y;                      // chunk
  const int b  = c >> 10;
  const int ch = c & 1023;
  size_t base = ((size_t)b * S_LEN + (size_t)k * CLEN) * D_DIM + ch;
  float A = 1.0f, h = 0.0f;
  for (int s = 0; s < CLEN; ++s) {
    size_t idx = base + (size_t)s * D_DIM;
    float av = a[idx];
    h = fmaf(av, h, xs[idx]);
    A *= av;
  }
  Aprod[(size_t)k * NCH + c] = A;
  Hend [(size_t)k * NCH + c] = h;
}

__global__ __launch_bounds__(256) void scan_pass2(
    const float* __restrict__ Aprod, const float* __restrict__ Hend,
    float* __restrict__ Hin) {
  const int c = blockIdx.x * 256 + threadIdx.x;
  float h = 0.0f;
  for (int k = 0; k < CHUNKS; ++k) {
    size_t idx = (size_t)k * NCH + c;
    Hin[idx] = h;
    h = fmaf(Aprod[idx], h, Hend[idx]);
  }
}

__global__ __launch_bounds__(256) void scan_pass3(
    float* __restrict__ xs, const float* __restrict__ a,
    const float* __restrict__ Hin) {
  const int c = blockIdx.x * 256 + threadIdx.x;
  const int k = blockIdx.y;
  const int b  = c >> 10;
  const int ch = c & 1023;
  float h = Hin[(size_t)k * NCH + c];
  size_t base = ((size_t)b * S_LEN + (size_t)k * CLEN) * D_DIM + ch;
  for (int s = 0; s < CLEN; ++s) {
    size_t idx = base + (size_t)s * D_DIM;
    h = fmaf(a[idx], h, xs[idx]);
    xs[idx] = h;   // in-place: xs becomes h
  }
}

extern "C" void kernel_launch(void* const* d_in, const int* in_sizes, int n_in,
                              void* d_out, int out_size, void* d_ws, size_t ws_size,
                              hipStream_t stream) {
  const float* x  = (const float*)d_in[0];
  const float* Wg = (const float*)d_in[1];
  const float* bg = (const float*)d_in[2];
  const float* Wv = (const float*)d_in[3];
  const float* bv = (const float*)d_in[4];
  const float* Wd = (const float*)d_in[5];
  const float* bd = (const float*)d_in[6];

  float* xs = (float*)d_out;                       // x_scan, then h in-place

  // workspace layout
  float* a     = (float*)d_ws;                     // 16M floats (64 MB)
  float* Aprod = a + (size_t)M_DIM * D_DIM;        // 256K floats
  float* Hend  = Aprod + (size_t)CHUNKS * NCH;     // 256K floats
  float* Hin   = Hend  + (size_t)CHUNKS * NCH;     // 256K floats
  unsigned short* xbf = (unsigned short*)(Hin + (size_t)CHUNKS * NCH); // 16M bf16
  unsigned short* wbf = xbf + (size_t)M_DIM * D_DIM;                   // 3M bf16

  // one-time bf16 conversion of activations and weights
  const int xn8 = (M_DIM * D_DIM) / 8;             // 2,097,152
  const int wn8 = (D_DIM * D_DIM) / 8;             // 131,072
  to_bf16<<<xn8 / 256, 256, 0, stream>>>(x, xbf, xn8);
  to_bf16<<<wn8 / 256, 256, 0, stream>>>(Wg, wbf + 0 * (size_t)D_DIM * D_DIM, wn8);
  to_bf16<<<wn8 / 256, 256, 0, stream>>>(Wv, wbf + 1 * (size_t)D_DIM * D_DIM, wn8);
  to_bf16<<<wn8 / 256, 256, 0, stream>>>(Wd, wbf + 2 * (size_t)D_DIM * D_DIM, wn8);

  dim3 ggrid(M_DIM / BM, D_DIM / BN);              // 256 x 16 blocks
  fused_gvd_gemm<<<ggrid, 256, 0, stream>>>(xbf, wbf, bg, bv, bd, xs, a);

  dim3 sgrid(NCH / 256, CHUNKS);                   // 16 x 64 blocks
  scan_pass1<<<sgrid, 256, 0, stream>>>(xs, a, Aprod, Hend);
  scan_pass2<<<NCH / 256, 256, 0, stream>>>(Aprod, Hend, Hin);
  scan_pass3<<<sgrid, 256, 0, stream>>>(xs, a, Hin);
}
